// TransformerBlock_17867063951393
// MI455X (gfx1250) — matrix operand order repaired and checked by reading, not compiled, not run on hardware
//
#include <hip/hip_runtime.h>

// ---------------------------------------------------------------------------
// Problem constants (match reference)
// ---------------------------------------------------------------------------
static constexpr int Bz   = 8;
static constexpr int Sq   = 1024;
static constexpr int Dm   = 1024;
static constexpr int Hh   = 16;
static constexpr int DKv  = 64;
static constexpr int Mrows = Bz * Sq;          // 8192

typedef __bf16 v16bf __attribute__((ext_vector_type(16)));
typedef float  v8f   __attribute__((ext_vector_type(8)));
typedef unsigned int u32x4 __attribute__((ext_vector_type(4)));
typedef int          i32x4 __attribute__((ext_vector_type(4)));
typedef int          i32x8 __attribute__((ext_vector_type(8)));

#define WMMA_BF16(a, b, c) \
  __builtin_amdgcn_wmma_f32_16x16x32_bf16(false, (a), false, (b), (short)0, (c), false, false)

#if __has_builtin(__builtin_amdgcn_tensor_load_to_lds) && \
    __has_builtin(__builtin_amdgcn_s_wait_tensorcnt)
#define USE_TDM 1
#else
#define USE_TDM 0
#endif

// RNE float -> bf16 in pure integer ops
__device__ __forceinline__ __bf16 f2bf(float f) {
  unsigned int u = __builtin_bit_cast(unsigned int, f);
  u += 0x7FFFu + ((u >> 16) & 1u);
  unsigned short h = (unsigned short)(u >> 16);
  return __builtin_bit_cast(__bf16, h);
}

// ---------------------------------------------------------------------------
// Vectorized fragment builders: two 16B LDS loads -> one 32B fragment.
// A (16x32 bf16, row-major LDS [row][k]); B (32x16, column-major LDS [n][k]).
// C/D (16x16 f32): elem i -> M = (lane>=16 ? 8:0)+i, N = lane&15
// ---------------------------------------------------------------------------
union FragU { uint4 u[2]; v16bf v; __bf16 h[16]; };

__device__ __forceinline__ v16bf frag_2x8(const __bf16* p0, const __bf16* p1) {
  FragU t;
  t.u[0] = *reinterpret_cast<const uint4*>(p0);
  t.u[1] = *reinterpret_cast<const uint4*>(p1);
  return t.v;
}

__device__ __forceinline__ v16bf load_a_frag(const __bf16* smem, int row0, int k0,
                                             int ldm, int lane) {
  const int h = (lane >> 4) & 1;
  const int r = lane & 15;
  const __bf16* p = smem + (row0 + r) * ldm + k0 + h * 8;
  return frag_2x8(p, p + 16);
}

__device__ __forceinline__ v16bf load_b_frag_cm(const __bf16* smem, int k0, int n0,
                                                int ldk, int lane) {
  const int h = (lane >> 4) & 1;
  const int r = lane & 15;
  const __bf16* p = smem + (n0 + r) * ldk + k0 + h * 8;
  return frag_2x8(p, p + 16);
}

#if USE_TDM
// ---------------------------------------------------------------------------
// TDM: one wave-level DMA loads a [rows x 32] bf16 tile (row stride = strideElems)
// into LDS with 32B padding after every 64B row -> LDS pitch 96B (LDK=48 halves).
// Descriptor packing per CDNA5 ISA §8.3 (group0) / §8.4 (group1).
// ---------------------------------------------------------------------------
__device__ __forceinline__ void tdm_load_tile32(const __bf16* gsrc, unsigned ldsOff,
                                                unsigned rows, unsigned strideElems,
                                                unsigned tensorRows) {
  const unsigned long long ga = (unsigned long long)(const void*)gsrc;
  u32x4 g0;
  g0[0] = 1u;                                        // count=1, user descriptor
  g0[1] = ldsOff;                                    // lds_addr (bytes)
  g0[2] = (unsigned)(ga & 0xFFFFFFFFull);            // global_addr[31:0]
  g0[3] = (unsigned)((ga >> 32) & 0x1FFFFFFull)      // global_addr[56:32]
          | (2u << 30);                              // type = 2 ("image")
  i32x8 g1;
  // word0: workgroup_mask=0 | data_size=1(2B)<<16 | pad_enable<<20
  //        | pad_interval=3 (64B)<<22 | pad_amount=7 (8 dwords=32B)<<25
  g1[0] = (int)((1u << 16) | (1u << 20) | (3u << 22) | (7u << 25));
  // word1: atomic_barrier_addr[15:0]=0 | tensor_dim0[15:0]<<16
  g1[1] = (int)((strideElems & 0xFFFFu) << 16);
  // word2: tensor_dim0[31:16] | tensor_dim1[15:0]<<16
  g1[2] = (int)(((strideElems >> 16) & 0xFFFFu) | ((tensorRows & 0xFFFFu) << 16));
  // word3: tensor_dim1[31:16] | tile_dim0(=32)<<16
  g1[3] = (int)(((tensorRows >> 16) & 0xFFFFu) | (32u << 16));
  // word4: tile_dim1(=rows) | tile_dim2(=0)<<16
  g1[4] = (int)(rows & 0xFFFFu);
  // word5/6: tensor_dim0_stride (48b, elems); tensor_dim1_stride = 0 (2D)
  g1[5] = (int)strideElems;
  g1[6] = 0;
  g1[7] = 0;
  const i32x4 z4 = {0, 0, 0, 0};
#if defined(__clang_major__) && (__clang_major__ >= 23)
  const i32x8 z8 = {0, 0, 0, 0, 0, 0, 0, 0};
  __builtin_amdgcn_tensor_load_to_lds(g0, g1, z4, z4, z8, 0);
#else
  __builtin_amdgcn_tensor_load_to_lds(g0, g1, z4, z4, 0);
#endif
}
#endif  // USE_TDM

// ---------------------------------------------------------------------------
// f32 -> bf16 elementwise conversion
// ---------------------------------------------------------------------------
__global__ __launch_bounds__(256) void cvt_kernel(const float* __restrict__ x,
                                                  __bf16* __restrict__ y, int n) {
  int i = blockIdx.x * 256 + threadIdx.x;
  if (i < n) y[i] = f2bf(x[i]);
}

// ---------------------------------------------------------------------------
// Tiled f32 [K,N] -> bf16 [N,K] transpose-convert (for W_q/W_k/W_v/W_o).
// ---------------------------------------------------------------------------
__global__ __launch_bounds__(256) void cvt_t_kernel(const float* __restrict__ x,
                                                    __bf16* __restrict__ y) {
  __shared__ __bf16 t[32][36];
  const int bn = blockIdx.x * 32;  // output row (n), input col
  const int bk = blockIdx.y * 32;  // output col (k), input row
  const int r  = threadIdx.x >> 3;
  const int c4 = (threadIdx.x & 7) * 4;
  const float4 v = *reinterpret_cast<const float4*>(x + (size_t)(bk + r) * Dm + bn + c4);
  t[c4 + 0][r] = f2bf(v.x);
  t[c4 + 1][r] = f2bf(v.y);
  t[c4 + 2][r] = f2bf(v.z);
  t[c4 + 3][r] = f2bf(v.w);
  __syncthreads();
  union { uint2 u; __bf16 h[4]; } o;
#pragma unroll
  for (int e = 0; e < 4; ++e) o.h[e] = t[r][c4 + e];
  *reinterpret_cast<uint2*>(y + (size_t)(bn + r) * Dm + bk + c4) = o.u;
}

// ---------------------------------------------------------------------------
// Tiled bf16 WMMA GEMM: C[M,N] = A[M,K] * B^T where B is stored [N,K].
// Double-buffered: TDM (wave 0 issues async tile DMA) when available,
// register-staged global loads otherwise. One barrier per k-step.
// Block tile 128x128x32, 256 threads (8 waves), wave tile 32x64 (2x4 accums).
// ---------------------------------------------------------------------------
template <bool BIAS, bool GELU, bool RES, bool OUTF32, bool TRV>
__global__ __launch_bounds__(256) void gemm_bf16_kernel(
    const __bf16* __restrict__ A, const __bf16* __restrict__ Bt,
    const float* __restrict__ bias, const float* __restrict__ res,
    float* __restrict__ Cf, __bf16* __restrict__ Cbf,
    int Md, int Nd, int Kd) {
  (void)Md;
  constexpr int BM = 128, BN = 128, BK = 32;
  constexpr int LDK = BK + 16;   // 48 halves = 96B LDS row pitch
  __shared__ __bf16 As[2][BM * LDK];    // row-major  [m][k]
  __shared__ __bf16 Bs[2][BN * LDK];    // col-major  [n][k]

  const int tid  = threadIdx.x;
  const int lane = tid & 31;
  const int w    = tid >> 5;
  const int wm   = (w >> 1) * 32;   // 0,32,64,96
  const int wn   = (w & 1) * 64;    // 0,64
  const int m0   = blockIdx.y * BM;
  const int n0   = blockIdx.x * BN;

  v8f acc[2][4];
#pragma unroll
  for (int i = 0; i < 2; ++i)
#pragma unroll
    for (int j = 0; j < 4; ++j) acc[i][j] = (v8f){0, 0, 0, 0, 0, 0, 0, 0};

  auto compute_tiles = [&](const __bf16* AsP, const __bf16* BsP) {
    v16bf af0 = load_a_frag(AsP, wm +  0, 0, LDK, lane);
    v16bf af1 = load_a_frag(AsP, wm + 16, 0, LDK, lane);
    v16bf bfj[4];
#pragma unroll
    for (int j = 0; j < 4; ++j) bfj[j] = load_b_frag_cm(BsP, 0, wn + j * 16, LDK, lane);
#pragma unroll
    for (int j = 0; j < 4; ++j) {
      acc[0][j] = WMMA_BF16(af0, bfj[j], acc[0][j]);
      acc[1][j] = WMMA_BF16(af1, bfj[j], acc[1][j]);
    }
  };

  const int T = Kd / BK;

#if USE_TDM
  const unsigned ldsA0 = (unsigned)(size_t)(&As[0][0]);
  const unsigned ldsA1 = (unsigned)(size_t)(&As[1][0]);
  const unsigned ldsB0 = (unsigned)(size_t)(&Bs[0][0]);
  const unsigned ldsB1 = (unsigned)(size_t)(&Bs[1][0]);
  if (w == 0) {  // prologue: DMA tile 0 into buffer 0
    tdm_load_tile32(A  + (size_t)m0 * Kd, ldsA0, BM, (unsigned)Kd, (unsigned)Mrows);
    tdm_load_tile32(Bt + (size_t)n0 * Kd, ldsB0, BN, (unsigned)Kd, (unsigned)Nd);
    __builtin_amdgcn_s_wait_tensorcnt(0);
  }
  __syncthreads();
  for (int it = 0; it < T; ++it) {
    const int cur = it & 1;
    if (w == 0 && it + 1 < T) {  // async DMA next tile into alternate buffer
      const size_t koff = (size_t)(it + 1) * BK;
      tdm_load_tile32(A  + (size_t)m0 * Kd + koff, cur ? ldsA0 : ldsA1,
                      BM, (unsigned)Kd, (unsigned)Mrows);
      tdm_load_tile32(Bt + (size_t)n0 * Kd + koff, cur ? ldsB0 : ldsB1,
                      BN, (unsigned)Kd, (unsigned)Nd);
    }
    compute_tiles(&As[cur][0], &Bs[cur][0]);      // overlaps with TDM
    if (w == 0 && it + 1 < T) __builtin_amdgcn_s_wait_tensorcnt(0);
    __syncthreads();
  }
#else
  // Register-staged double buffering fallback.
  const int tr = tid >> 1, tc = (tid & 1) * 16;   // 128 rows x 32 k
  const __bf16* aptr = A  + (size_t)(m0 + tr) * Kd + tc;
  const __bf16* bptr = Bt + (size_t)(n0 + tr) * Kd + tc;
  uint4 ra0, ra1, rb0, rb1;
  auto gload = [&](int k0) {
    const uint4* s4a = reinterpret_cast<const uint4*>(aptr + k0);
    ra0 = s4a[0]; ra1 = s4a[1];
    const uint4* s4b = reinterpret_cast<const uint4*>(bptr + k0);
    rb0 = s4b[0]; rb1 = s4b[1];
    __builtin_prefetch(aptr + k0 + BK, 0, 1);
    __builtin_prefetch(bptr + k0 + BK, 0, 1);
  };
  auto lstore = [&](int buf) {
    uint4* d4a = reinterpret_cast<uint4*>(&As[buf][tr * LDK + tc]);
    d4a[0] = ra0; d4a[1] = ra1;
    uint4* d4b = reinterpret_cast<uint4*>(&Bs[buf][tr * LDK + tc]);
    d4b[0] = rb0; d4b[1] = rb1;
  };
  gload(0); lstore(0);
  if (T > 1) gload(BK);
  __syncthreads();
  for (int it = 0; it < T; ++it) {
    const int cur = it & 1;
    compute_tiles(&As[cur][0], &Bs[cur][0]);
    if (it + 1 < T) {
      lstore(1 - cur);
      if (it + 2 < T) gload((it + 2) * BK);
    }
    __syncthreads();
  }
#endif

  // Epilogue
  const int h = (lane >> 4) & 1;
  const int r = lane & 15;
#pragma unroll
  for (int i = 0; i < 2; ++i) {
#pragma unroll
    for (int j = 0; j < 4; ++j) {
#pragma unroll
      for (int e = 0; e < 8; ++e) {
        const int row = m0 + wm + i * 16 + h * 8 + e;
        const int col = n0 + wn + j * 16 + r;
        float v = acc[i][j][e];
        if (BIAS) v += bias[col];
        if (GELU) v = 0.5f * v * (1.0f + erff(v * 0.70710678118654752f));
        if (RES)  v += res[(size_t)row * Nd + col];
        if (OUTF32) {
          Cf[(size_t)row * Nd + col] = v;
        } else if (TRV) {
          const size_t o = ((size_t)((row >> 10) * Dm + col) << 10) + (row & 1023);
          Cbf[o] = f2bf(v);
        } else {
          Cbf[(size_t)row * Nd + col] = f2bf(v);
        }
      }
    }
  }
}

// ---------------------------------------------------------------------------
// Causal scores: per (b,h) 64-row tile, scores = q * k^T * scale,
// writes raw masked scores (0 above diagonal) to att (f32).
// ---------------------------------------------------------------------------
__global__ __launch_bounds__(256) void scores_kernel(const __bf16* __restrict__ qb,
                                                     const __bf16* __restrict__ kb,
                                                     float* __restrict__ att) {
  const int bh = blockIdx.x;         // b*H + h
  const int b  = bh >> 4;
  const int hh = bh & 15;
  const int ty = blockIdx.y;         // 64-row tile, 16 tiles
  constexpr int LQ = DKv + 16;       // 80 halves = 160B rows
  __shared__ __bf16 Qs[64 * LQ];     // [row][k]
  __shared__ __bf16 Ks[64 * LQ];     // [n][k]

  const int tid = threadIdx.x, lane = tid & 31, w = tid >> 5;
  const int qr = tid >> 2, qc = (tid & 3) * 16;
  {  // q tile 64 rows x 64 dk
    const __bf16* src = qb + ((size_t)(b * Sq + ty * 64 + qr)) * Dm + hh * DKv + qc;
    const uint4* s4 = reinterpret_cast<const uint4*>(src);
    uint4 u0 = s4[0], u1 = s4[1];
    uint4* d4 = reinterpret_cast<uint4*>(&Qs[qr * LQ + qc]);
    d4[0] = u0; d4[1] = u1;
  }
  __syncthreads();

  const float scale = 1.0f / (8.0f + 1e-6f);   // 1/(sqrt(64)+eps)
  const int wm = (w >> 1) * 16;  // 4 row groups
  const int wn = (w & 1) * 32;   // 2 col groups
  const size_t attBase = (size_t)bh * Sq * Sq;
  const int h = (lane >> 4) & 1, r = lane & 15;

  for (int ct = 0; ct <= ty; ++ct) {
    {  // K tile: Ks[n][k] = kb[(b*S + ct*64+n)*D + h*64 + k]  (direct copy)
      const __bf16* src = kb + ((size_t)(b * Sq + ct * 64 + qr)) * Dm + hh * DKv + qc;
      const uint4* s4 = reinterpret_cast<const uint4*>(src);
      uint4 u0 = s4[0], u1 = s4[1];
      uint4* d4 = reinterpret_cast<uint4*>(&Ks[qr * LQ + qc]);
      d4[0] = u0; d4[1] = u1;
    }
    __syncthreads();

    v8f acc[2];
    acc[0] = (v8f){0, 0, 0, 0, 0, 0, 0, 0};
    acc[1] = (v8f){0, 0, 0, 0, 0, 0, 0, 0};
#pragma unroll
    for (int ks = 0; ks < DKv; ks += 32) {
      v16bf a = load_a_frag(Qs, wm, ks, LQ, lane);
#pragma unroll
      for (int j = 0; j < 2; ++j) {
        v16bf bb = load_b_frag_cm(Ks, ks, wn + j * 16, LQ, lane);
        acc[j] = WMMA_BF16(a, bb, acc[j]);
      }
    }
#pragma unroll
    for (int j = 0; j < 2; ++j) {
#pragma unroll
      for (int e = 0; e < 8; ++e) {
        const int gr = ty * 64 + wm + h * 8 + e;
        const int gc = ct * 64 + wn + j * 16 + r;
        const float v = (gc <= gr) ? acc[j][e] * scale : 0.0f;
        att[attBase + (size_t)gr * Sq + gc] = v;
      }
    }
    __syncthreads();
  }
  // zero the fully-masked upper-triangle column tiles
  for (int ct = ty + 1; ct < 16; ++ct) {
    const int r2 = tid >> 2, c0 = (tid & 3) * 16;
    float* dst = att + attBase + (size_t)(ty * 64 + r2) * Sq + ct * 64 + c0;
    const float4 z = {0.f, 0.f, 0.f, 0.f};
#pragma unroll
    for (int q4 = 0; q4 < 4; ++q4) reinterpret_cast<float4*>(dst)[q4] = z;
  }
}

// ---------------------------------------------------------------------------
// Row softmax over causal prefix (cols 0..s); cols > s stay 0.
// ---------------------------------------------------------------------------
__global__ __launch_bounds__(256) void softmax_kernel(float* __restrict__ att) {
  const int row = blockIdx.x;        // 0 .. B*H*S-1
  const int s   = row & (Sq - 1);
  float* p = att + (size_t)row * Sq;
  const int tid = threadIdx.x;
  __shared__ float red[256];

  float m = -3.402823466e38f;
  for (int c = tid; c < Sq; c += 256)
    if (c <= s) m = fmaxf(m, p[c]);
  red[tid] = m;
  __syncthreads();
  for (int st = 128; st > 0; st >>= 1) {
    if (tid < st) red[tid] = fmaxf(red[tid], red[tid + st]);
    __syncthreads();
  }
  m = red[0];
  __syncthreads();

  float lsum = 0.f;
  for (int c = tid; c < Sq; c += 256) {
    if (c <= s) {
      const float e = __expf(p[c] - m);
      p[c] = e;
      lsum += e;
    }
  }
  red[tid] = lsum;
  __syncthreads();
  for (int st = 128; st > 0; st >>= 1) {
    if (tid < st) red[tid] += red[tid + st];
    __syncthreads();
  }
  const float inv = 1.0f / red[0];
  for (int c = tid; c < Sq; c += 256)
    if (c <= s) p[c] *= inv;
}

// ---------------------------------------------------------------------------
// v_att = att @ v using head-transposed vT[(b*H+h)*64+dv][s] (bf16).
// ---------------------------------------------------------------------------
__global__ __launch_bounds__(256) void av_kernel(const float* __restrict__ att,
                                                 const __bf16* __restrict__ vT,
                                                 __bf16* __restrict__ concat) {
  const int bh = blockIdx.x, b = bh >> 4, hh = bh & 15;
  const int ty = blockIdx.y;                 // 128-row tile, 8 tiles
  constexpr int LDK = 48;                    // 32 k + pad
  __shared__ __bf16 Ats[128 * LDK];          // [row][k]
  __shared__ __bf16 Vs[64 * LDK];            // [n(dv)][k]

  const int tid = threadIdx.x, lane = tid & 31, w = tid >> 5;
  const int rowBase = ty * 128;
  const size_t attBase = (size_t)bh * Sq * Sq;

  v8f acc[4];
#pragma unroll
  for (int j = 0; j < 4; ++j) acc[j] = (v8f){0, 0, 0, 0, 0, 0, 0, 0};

  const int ar = tid >> 1, ac = (tid & 1) * 16;   // att tile 128x32
  const int vn = tid >> 2, vc = (tid & 3) * 8;    // v tile 64x32
  const int ktiles = (ty + 1) * 4;                // causal bound
  for (int kt = 0; kt < ktiles; ++kt) {
    const int k0 = kt * 32;
    {  // att tile f32 -> bf16, packed store
      const float* src = att + attBase + (size_t)(rowBase + ar) * Sq + k0 + ac;
      FragU t;
#pragma unroll
      for (int e = 0; e < 16; ++e) t.h[e] = f2bf(src[e]);
      uint4* d4 = reinterpret_cast<uint4*>(&Ats[ar * LDK + ac]);
      d4[0] = t.u[0]; d4[1] = t.u[1];
    }
    {  // v tile: Vs[n][k] = vT[(bh*64+n)*S + k]  (direct vectorized copy)
      const __bf16* src = vT + ((size_t)(bh * 64 + vn)) * Sq + k0 + vc;
      *reinterpret_cast<uint4*>(&Vs[vn * LDK + vc]) =
          *reinterpret_cast<const uint4*>(src);
    }
    __syncthreads();

    v16bf a = load_a_frag(Ats, w * 16, 0, LDK, lane);
#pragma unroll
    for (int j = 0; j < 4; ++j) {
      v16bf bb = load_b_frag_cm(Vs, 0, j * 16, LDK, lane);
      acc[j] = WMMA_BF16(a, bb, acc[j]);
    }
    __syncthreads();
  }

  const int h = (lane >> 4) & 1, r = lane & 15;
#pragma unroll
  for (int j = 0; j < 4; ++j) {
#pragma unroll
    for (int e = 0; e < 8; ++e) {
      const int row = rowBase + w * 16 + h * 8 + e;
      const int col = j * 16 + r;
      concat[((size_t)(b * Sq + row)) * Dm + hh * DKv + col] = f2bf(acc[j][e]);
    }
  }
}

// ---------------------------------------------------------------------------
// LayerNorm over D=1024; writes f32 (optional) and bf16 (optional) outputs.
// ---------------------------------------------------------------------------
__global__ __launch_bounds__(256) void layernorm_kernel(
    const float* __restrict__ x, const float* __restrict__ g,
    const float* __restrict__ bta, float* __restrict__ yf,
    __bf16* __restrict__ ybf) {
  __shared__ float r1[256];
  __shared__ float r2[256];
  const int tid = threadIdx.x;
  const float* p = x + (size_t)blockIdx.x * Dm;
  float vals[4];
  float s = 0.f, s2 = 0.f;
#pragma unroll
  for (int i = 0; i < 4; ++i) {
    const float v = p[tid + 256 * i];
    vals[i] = v; s += v; s2 += v * v;
  }
  r1[tid] = s; r2[tid] = s2;
  __syncthreads();
  for (int st = 128; st > 0; st >>= 1) {
    if (tid < st) { r1[tid] += r1[tid + st]; r2[tid] += r2[tid + st]; }
    __syncthreads();
  }
  const float mean = r1[0] * (1.0f / 1024.0f);
  const float var  = r2[0] * (1.0f / 1024.0f) - mean * mean;
  const float rstd = rsqrtf(var + 1e-5f);
#pragma unroll
  for (int i = 0; i < 4; ++i) {
    const int c = tid + 256 * i;
    const float o = g[c] * (vals[i] - mean) * rstd + bta[c];
    if (yf)  yf[(size_t)blockIdx.x * Dm + c] = o;
    if (ybf) ybf[(size_t)blockIdx.x * Dm + c] = f2bf(o);
  }
}

// ---------------------------------------------------------------------------
// Host orchestration
// ---------------------------------------------------------------------------
extern "C" void kernel_launch(void* const* d_in, const int* in_sizes, int n_in,
                              void* d_out, int out_size, void* d_ws, size_t ws_size,
                              hipStream_t stream) {
  (void)in_sizes; (void)n_in; (void)out_size; (void)ws_size;
  const float* Q    = (const float*)d_in[0];
  const float* Kin  = (const float*)d_in[1];
  const float* Vin  = (const float*)d_in[2];
  const float* W_q  = (const float*)d_in[4];
  const float* W_k  = (const float*)d_in[5];
  const float* W_v  = (const float*)d_in[6];
  const float* W_o  = (const float*)d_in[7];
  const float* w1   = (const float*)d_in[8];
  const float* b1   = (const float*)d_in[9];
  const float* w2   = (const float*)d_in[10];
  const float* b2   = (const float*)d_in[11];
  const float* ln_g = (const float*)d_in[12];
  const float* ln_b = (const float*)d_in[13];
  // d_in[3] (mask) is all-false: ignored.

  float* out = (float*)d_out;                        // [B,S,D]
  float* att = out + (size_t)Mrows * Dm;             // [B*H,S,S]

  const size_t MD = (size_t)Mrows * Dm;              // 8388608
  const size_t DD = (size_t)Dm * Dm;                 // 1048576
  char* ws = (char*)d_ws;
  auto alloc_bf = [&](size_t n) { __bf16* p = (__bf16*)ws; ws += n * 2; return p; };
  auto alloc_f  = [&](size_t n) { float*  p = (float*)ws;  ws += n * 4; return p; };

  __bf16* Qbf  = alloc_bf(MD);
  __bf16* Kbf  = alloc_bf(MD);
  __bf16* Vbf  = alloc_bf(MD);
  __bf16* WqT  = alloc_bf(DD);   // [N,K] transposed weights
  __bf16* WkT  = alloc_bf(DD);
  __bf16* WvT  = alloc_bf(DD);
  __bf16* WoT  = alloc_bf(DD);
  __bf16* w1bf = alloc_bf(DD);   // already [N,K] semantics (X @ w1.T)
  __bf16* w2bf = alloc_bf(DD);
  __bf16* qb   = alloc_bf(MD);   // [B*S, H*DK]
  __bf16* kb   = alloc_bf(MD);
  __bf16* vT   = alloc_bf(MD);   // [(b*H+h)*64+dv][s]
  __bf16* cat  = alloc_bf(MD);
  float*  ares = alloc_f(MD);    // attn_out + Q
  float*  Xf   = alloc_f(MD);
  __bf16* Xbf  = alloc_bf(MD);
  __bf16* H1   = alloc_bf(MD);
  float*  Yf   = alloc_f(MD);

  auto cvt = [&](const float* x, __bf16* y, size_t n) {
    cvt_kernel<<<(unsigned)((n + 255) / 256), 256, 0, stream>>>(x, y, (int)n);
  };
  const dim3 gT(Dm / 32, Dm / 32);
  cvt(Q, Qbf, MD);  cvt(Kin, Kbf, MD);  cvt(Vin, Vbf, MD);
  cvt_t_kernel<<<gT, 256, 0, stream>>>(W_q, WqT);
  cvt_t_kernel<<<gT, 256, 0, stream>>>(W_k, WkT);
  cvt_t_kernel<<<gT, 256, 0, stream>>>(W_v, WvT);
  cvt_t_kernel<<<gT, 256, 0, stream>>>(W_o, WoT);
  cvt(w1, w1bf, DD);  cvt(w2, w2bf, DD);

  const dim3 g8(Dm / 128, Mrows / 128);  // (8, 64)

  // QKV projections
  gemm_bf16_kernel<false, false, false, false, false>
      <<<g8, 256, 0, stream>>>(Qbf, WqT, nullptr, nullptr, nullptr, qb, Mrows, Dm, Dm);
  gemm_bf16_kernel<false, false, false, false, false>
      <<<g8, 256, 0, stream>>>(Kbf, WkT, nullptr, nullptr, nullptr, kb, Mrows, Dm, Dm);
  gemm_bf16_kernel<false, false, false, false, true>   // head-transposed V output
      <<<g8, 256, 0, stream>>>(Vbf, WvT, nullptr, nullptr, nullptr, vT, Mrows, Dm, Dm);

  // scores (scaled, causal-masked, raw) -> att ; softmax in place ; att @ v
  scores_kernel<<<dim3(Bz * Hh, Sq / 64), 256, 0, stream>>>(qb, kb, att);
  softmax_kernel<<<Bz * Hh * Sq, 256, 0, stream>>>(att);
  av_kernel<<<dim3(Bz * Hh, Sq / 128), 256, 0, stream>>>(att, vT, cat);

  // attn_out = concat @ W_o + Q (residual), f32
  gemm_bf16_kernel<false, false, true, true, false>
      <<<g8, 256, 0, stream>>>(cat, WoT, nullptr, Q, ares, nullptr, Mrows, Dm, Dm);
  // X = LN(attn_out + Q) -> f32 + bf16
  layernorm_kernel<<<Mrows, 256, 0, stream>>>(ares, ln_g, ln_b, Xf, Xbf);

  // H1 = gelu(X @ w1^T + b1) -> bf16
  gemm_bf16_kernel<true, true, false, false, false>
      <<<g8, 256, 0, stream>>>(Xbf, w1bf, b1, nullptr, nullptr, H1, Mrows, Dm, Dm);
  // Y = H1 @ w2^T + b2 + X -> f32
  gemm_bf16_kernel<true, false, true, true, false>
      <<<g8, 256, 0, stream>>>(H1, w2bf, b2, Xf, Yf, nullptr, Mrows, Dm, Dm);
  // output = LN(Y)
  layernorm_kernel<<<Mrows, 256, 0, stream>>>(Yf, ln_g, ln_b, out, nullptr);
}
